// _S4DLayer_73693048865111
// MI455X (gfx1250) — matrix operand order repaired and checked by reading, not compiled, not run on hardware
//
#include <hip/hip_runtime.h>

// ---------------------------------------------------------------------------
// S4D layer for MI455X (gfx1250, wave32, WMMA + TDM).
//   prep: B_w / C_w / D_w  -> bf16 copies in ws (once, L2-resident)
//   Bu  = u @ B_w^T            WMMA bf16, weights staged via tensor_load_to_lds
//   h   = chunked parallel scan of diagonal complex recurrence
//   y   = h_cat @ C_w^T + u @ D_w^T   WMMA bf16, TDM-staged weights
// ---------------------------------------------------------------------------

typedef __attribute__((ext_vector_type(16))) __bf16   v16bf;
typedef __attribute__((ext_vector_type(8)))  float    v8f;
typedef __attribute__((ext_vector_type(4)))  unsigned u32x4;
typedef __attribute__((ext_vector_type(8)))  int      i32x8;
typedef __attribute__((ext_vector_type(4)))  int      i32x4;

#define D_MODEL 512
#define D_STATE 64
#define NPROJ   128           // 2 * D_STATE
#define BATCH   8
#define SEQ     4096
#define M_TOTAL (BATCH * SEQ) // 32768
#define CHUNK   128
#define NCHUNK  (SEQ / CHUNK) // 32
#define DT_HALF 0.5f          // DT/2 with DT = 1.0

// ---------------------------------------------------------------------------
// Bilinear-discretized diagonal A_bar (complex), per state n.
// ---------------------------------------------------------------------------
__device__ __forceinline__ void compute_Abar(const float* __restrict__ log_A_real,
                                             const float* __restrict__ log_A_imag,
                                             int n, float& Ar, float& Ai) {
    float ar = -__expf(log_A_real[n]);
    float ai = log_A_imag[n];
    float nr = 1.0f + DT_HALF * ar;
    float ni = DT_HALF * ai;
    float dr = 1.0f - DT_HALF * ar;
    float di = -DT_HALF * ai;
    float den = dr * dr + di * di;
    Ar = (nr * dr + ni * di) / den;
    Ai = (ni * dr - nr * di) / den;
}

// ---------------------------------------------------------------------------
// TDM: issue tensor_load_to_lds of a [128 rows x 32 elem] bf16 tile.
//   gsrc : global bf16 tile start, row stride ldw elements
//   D# per CDNA5 ISA 8.3/8.4: group0 = {count|lds_addr|global_addr|type=2},
//   group1 = {data_size=2B, tensor_dim0/1, tile_dim0=32, tile_dim1=128, stride0=ldw}.
//   2D tensor -> trailing groups zero. Tracked by TENSORcnt.
//   This toolchain exposes the 6-arg builtin form.
// ---------------------------------------------------------------------------
__device__ __forceinline__ void tdm_load_tile(unsigned lds_addr,
                                              const __bf16* __restrict__ gsrc,
                                              unsigned ldw, unsigned nrows) {
    unsigned long long ga = (unsigned long long)(uintptr_t)gsrc;
    u32x4 g0;
    g0[0] = 1u;                                    // count=1 (valid user D#)
    g0[1] = lds_addr;                              // lds_addr [63:32]
    g0[2] = (unsigned)(ga & 0xFFFFFFFFu);          // global_addr [95:64]
    g0[3] = (unsigned)((ga >> 32) & 0x01FFFFFFu)   // global_addr [120:96]
          | 0x80000000u;                           // type=2 ("image") [127:126]
    i32x8 g1;
    g1[0] = 0x00010000;                            // data_size=1 -> 2-byte elems
    g1[1] = (int)((ldw & 0xFFFFu) << 16);          // tensor_dim0[15:0]
    g1[2] = (int)((ldw >> 16) | ((nrows & 0xFFFFu) << 16)); // dim0[31:16]|dim1[15:0]
    g1[3] = (int)((nrows >> 16) | (32u << 16));    // dim1[31:16] | tile_dim0=32
    g1[4] = 128;                                   // tile_dim1=128, tile_dim2=0
    g1[5] = (int)ldw;                              // tensor_dim0_stride[31:0]
    g1[6] = 0;                                     // stride0[47:32] | stride1[15:0]
    g1[7] = 0;                                     // stride1[47:16]
    i32x4 z4 = {0, 0, 0, 0};
    i32x8 z8 = {0, 0, 0, 0, 0, 0, 0, 0};
    __builtin_amdgcn_tensor_load_to_lds(g0, g1, z4, z4, z8, 0);
}

// ---------------------------------------------------------------------------
// WMMA fragment loaders.
// A (16x32 bf16, MxK): lane holds row M = lane&15;
//   per-lane K groups [kb, kb+8) and [kb+16, kb+24), kb = k0 + ((lane&16)?8:0).
// ---------------------------------------------------------------------------
__device__ __forceinline__ v16bf load_a_frag(const float* __restrict__ rowp, int kb) {
    float4 x0 = *(const float4*)(rowp + kb);
    float4 x1 = *(const float4*)(rowp + kb + 4);
    float4 y0 = *(const float4*)(rowp + kb + 16);
    float4 y1 = *(const float4*)(rowp + kb + 20);
    v16bf a;
    a[0]  = (__bf16)x0.x; a[1]  = (__bf16)x0.y; a[2]  = (__bf16)x0.z; a[3]  = (__bf16)x0.w;
    a[4]  = (__bf16)x1.x; a[5]  = (__bf16)x1.y; a[6]  = (__bf16)x1.z; a[7]  = (__bf16)x1.w;
    a[8]  = (__bf16)y0.x; a[9]  = (__bf16)y0.y; a[10] = (__bf16)y0.z; a[11] = (__bf16)y0.w;
    a[12] = (__bf16)y1.x; a[13] = (__bf16)y1.y; a[14] = (__bf16)y1.z; a[15] = (__bf16)y1.w;
    return a;
}

// B (32x16 bf16, KxN): lane holds column N = lane&15; K groups [kb, kb+8) and
// [kb+16, kb+24), kb = (lane&16)?8:0.  LDS tile layout Bs[col][kk], kk in [0,32).
__device__ __forceinline__ v16bf load_b_frag(const __bf16* __restrict__ Bs, int ncol, int lane) {
    typedef __attribute__((ext_vector_type(8))) __bf16 v8bf;
    int n  = ncol + (lane & 15);
    int kb = (lane & 16) ? 8 : 0;
    v8bf lo = *(const v8bf*)(Bs + n * 32 + kb);
    v8bf hi = *(const v8bf*)(Bs + n * 32 + kb + 16);
    return __builtin_shufflevector(lo, hi, 0,1,2,3,4,5,6,7,8,9,10,11,12,13,14,15);
}

// 8 B fragments loaded up front, then 8 back-to-back WMMAs (no per-WMMA
// dscnt stall; RAW-free WMMA burst for the matrix pipe).
__device__ __forceinline__ void wmma_burst8(v8f (&acc)[8], v16bf a,
                                            const __bf16* __restrict__ Bsc, int lane) {
    v16bf bfrag[8];
#pragma unroll
    for (int nt = 0; nt < 8; ++nt) bfrag[nt] = load_b_frag(Bsc, nt * 16, lane);
#pragma unroll
    for (int nt = 0; nt < 8; ++nt)
        acc[nt] = __builtin_amdgcn_wmma_f32_16x16x32_bf16(
            false, a, false, bfrag[nt], (short)0, acc[nt], false, false);
}

// ---------------------------------------------------------------------------
// Prep: f32 -> bf16 weight conversion (one-time, tiny).
// ---------------------------------------------------------------------------
__global__ __launch_bounds__(256) void s4d_cvt_bf16(const float* __restrict__ src,
                                                    __bf16* __restrict__ dst, int n4) {
    int i = (blockIdx.x * 256 + threadIdx.x);
    if (i < n4) {
        float4 v = ((const float4*)src)[i];
        __bf16* d = dst + i * 4;
        d[0] = (__bf16)v.x; d[1] = (__bf16)v.y; d[2] = (__bf16)v.z; d[3] = (__bf16)v.w;
    }
}

// ---------------------------------------------------------------------------
// Kernel 1: Bu = u @ B_w^T.  256 blocks x 256 thr; block -> 128 rows x 128 cols.
// Weight tiles DMA'd by TDM, double-buffered against the WMMA pipe.
// ---------------------------------------------------------------------------
__global__ __launch_bounds__(256) void s4d_proj_in(const float* __restrict__ u,
                                                   const __bf16* __restrict__ Bwb,
                                                   float* __restrict__ Bu) {
    __shared__ __bf16 Bs[2][NPROJ * 32]; // 2 x 8 KB
    const int tid  = threadIdx.x;
    const int lane = tid & 31;
    const int wave = tid >> 5;
    const int m0   = blockIdx.x * 128 + wave * 16;
    const float* arow = u + (size_t)(m0 + (lane & 15)) * D_MODEL;
    const int akoff = (lane & 16) ? 8 : 0;
    const unsigned lds0 = (unsigned)(uintptr_t)&Bs[0][0];
    const unsigned lds1 = (unsigned)(uintptr_t)&Bs[1][0];

    v8f acc[8];
#pragma unroll
    for (int nt = 0; nt < 8; ++nt) acc[nt] = (v8f)(0.0f);

    if (tid < 32) tdm_load_tile(lds0, Bwb, D_MODEL, NPROJ);

    int cur = 0;
    for (int k0 = 0; k0 < D_MODEL; k0 += 32) {
        if (tid < 32) __builtin_amdgcn_s_wait_tensorcnt(0);
        __syncthreads();
        if (tid < 32 && k0 + 32 < D_MODEL)
            tdm_load_tile(cur ? lds0 : lds1, Bwb + (k0 + 32), D_MODEL, NPROJ);
        if (k0 + 32 < D_MODEL)
            __builtin_prefetch(arow + k0 + 32 + akoff, 0, 1); // global_prefetch_b8
        v16bf a = load_a_frag(arow, k0 + akoff);
        wmma_burst8(acc, a, Bs[cur], lane);
        cur ^= 1;
    }

    const int colbase = lane & 15;
    const int rowoff  = (lane & 16) ? 8 : 0;
#pragma unroll
    for (int nt = 0; nt < 8; ++nt)
#pragma unroll
        for (int r = 0; r < 8; ++r)
            Bu[(size_t)(m0 + rowoff + r) * NPROJ + nt * 16 + colbase] = acc[nt][r];
}

// ---------------------------------------------------------------------------
// Kernel 2a: local scans, h0 = 0, per (batch, chunk, state). 256 blocks x 64 thr.
// ---------------------------------------------------------------------------
__global__ __launch_bounds__(64) void s4d_scan_local(const float* __restrict__ lar,
                                                     const float* __restrict__ lai,
                                                     const float* __restrict__ Bu,
                                                     float* __restrict__ H,
                                                     float2* __restrict__ F) {
    const int c = blockIdx.x % NCHUNK;
    const int b = blockIdx.x / NCHUNK;
    const int n = threadIdx.x;
    float Ar, Ai;
    compute_Abar(lar, lai, n, Ar, Ai);

    float hr = 0.0f, hi = 0.0f;
    const int row0 = b * SEQ + c * CHUNK;
    for (int t = 0; t < CHUNK; ++t) {
        const float* bp = Bu + (size_t)(row0 + t) * NPROJ;
        float xr = bp[n], xi = bp[D_STATE + n];
        float nhr = Ar * hr - Ai * hi + xr;
        float nhi = Ar * hi + Ai * hr + xi;
        hr = nhr; hi = nhi;
        float* hp = H + (size_t)(row0 + t) * NPROJ;
        hp[n] = hr; hp[D_STATE + n] = hi;
    }
    F[(size_t)(b * D_STATE + n) * NCHUNK + c] = make_float2(hr, hi);
}

// ---------------------------------------------------------------------------
// Kernel 2b: sequential combine: Hpre[c] = A^CHUNK * Hpre[c-1] + F[c].
// ---------------------------------------------------------------------------
__global__ __launch_bounds__(64) void s4d_scan_combine(const float* __restrict__ lar,
                                                       const float* __restrict__ lai,
                                                       const float2* __restrict__ F,
                                                       float2* __restrict__ Hpre) {
    const int b = blockIdx.x;
    const int n = threadIdx.x;
    float Ar, Ai;
    compute_Abar(lar, lai, n, Ar, Ai);

    float alr = 1.0f, ali = 0.0f; // A^CHUNK
    for (int i = 0; i < CHUNK; ++i) {
        float tr = alr * Ar - ali * Ai;
        ali = alr * Ai + ali * Ar;
        alr = tr;
    }
    float hr = 0.0f, hi = 0.0f;
    const size_t base = (size_t)(b * D_STATE + n) * NCHUNK;
    for (int c = 0; c < NCHUNK; ++c) {
        float2 f = F[base + c];
        float tr = alr * hr - ali * hi + f.x;
        float ti = alr * hi + ali * hr + f.y;
        hr = tr; hi = ti;
        Hpre[base + c] = make_float2(hr, hi);
    }
}

// ---------------------------------------------------------------------------
// Kernel 2c: fix-up: h[t] += A^(t_local+1) * Hpre[c-1]. 256 blocks x 64 thr.
// ---------------------------------------------------------------------------
__global__ __launch_bounds__(64) void s4d_scan_fix(const float* __restrict__ lar,
                                                   const float* __restrict__ lai,
                                                   const float2* __restrict__ Hpre,
                                                   float* __restrict__ H) {
    const int c = blockIdx.x % NCHUNK;
    const int b = blockIdx.x / NCHUNK;
    if (c == 0) return;
    const int n = threadIdx.x;
    float Ar, Ai;
    compute_Abar(lar, lai, n, Ar, Ai);

    float2 h0 = Hpre[(size_t)(b * D_STATE + n) * NCHUNK + (c - 1)];
    float pr = Ar, pi = Ai; // A^1
    const int row0 = b * SEQ + c * CHUNK;
    for (int t = 0; t < CHUNK; ++t) {
        float* hp = H + (size_t)(row0 + t) * NPROJ;
        hp[n]           += pr * h0.x - pi * h0.y;
        hp[D_STATE + n] += pr * h0.y + pi * h0.x;
        float npr = pr * Ar - pi * Ai;
        float npi = pr * Ai + pi * Ar;
        pr = npr; pi = npi;
    }
}

// ---------------------------------------------------------------------------
// Kernel 3: y = H @ C_w^T + u @ D_w^T. grid (256, 4) x 256 thr; tile 128 x 128.
// One flattened 20-step pipeline: steps 0-3 over C_w (K=128, A from H),
// steps 4-19 over D_w (K=512, A from u). TDM double-buffers across phases.
// ---------------------------------------------------------------------------
__global__ __launch_bounds__(256) void s4d_proj_out(const float* __restrict__ H,
                                                    const float* __restrict__ u,
                                                    const __bf16* __restrict__ Cwb,
                                                    const __bf16* __restrict__ Dwb,
                                                    float* __restrict__ out) {
    __shared__ __bf16 Bs[2][128 * 32]; // 2 x 8 KB
    const int tid   = threadIdx.x;
    const int lane  = tid & 31;
    const int wave  = tid >> 5;
    const int m0    = blockIdx.x * 128 + wave * 16;
    const int n0blk = blockIdx.y * 128;
    const int arow_id = m0 + (lane & 15);
    const int akoff   = (lane & 16) ? 8 : 0;
    const float* hrow = H + (size_t)arow_id * NPROJ;
    const float* urow = u + (size_t)arow_id * D_MODEL;
    const unsigned lds0 = (unsigned)(uintptr_t)&Bs[0][0];
    const unsigned lds1 = (unsigned)(uintptr_t)&Bs[1][0];

    v8f acc[8];
#pragma unroll
    for (int nt = 0; nt < 8; ++nt) acc[nt] = (v8f)(0.0f);

    // step s -> weight tile source (global bf16 ptr) for the TDM
    auto tile_src = [&](int s) -> const __bf16* {
        if (s < 4) return Cwb + (size_t)n0blk * NPROJ + s * 32;
        return Dwb + (size_t)n0blk * D_MODEL + (s - 4) * 32;
    };

    if (tid < 32) tdm_load_tile(lds0, tile_src(0), NPROJ, 128);

    int cur = 0;
    for (int s = 0; s < 20; ++s) {
        if (tid < 32) __builtin_amdgcn_s_wait_tensorcnt(0);
        __syncthreads();
        if (tid < 32 && s + 1 < 20) {
            unsigned ldw = (s + 1 < 4) ? NPROJ : D_MODEL;
            tdm_load_tile(cur ? lds0 : lds1, tile_src(s + 1), ldw, 128);
        }
        v16bf a;
        if (s < 4) {
            a = load_a_frag(hrow, s * 32 + akoff);
        } else {
            int k0 = (s - 4) * 32;
            if (k0 + 32 < D_MODEL)
                __builtin_prefetch(urow + k0 + 32 + akoff, 0, 1);
            a = load_a_frag(urow, k0 + akoff);
        }
        wmma_burst8(acc, a, Bs[cur], lane);
        cur ^= 1;
    }

    const int colbase = lane & 15;
    const int rowoff  = (lane & 16) ? 8 : 0;
#pragma unroll
    for (int nt = 0; nt < 8; ++nt)
#pragma unroll
        for (int r = 0; r < 8; ++r)
            out[(size_t)(m0 + rowoff + r) * D_MODEL + n0blk + nt * 16 + colbase] = acc[nt][r];
}

// ---------------------------------------------------------------------------
// Host launcher.
// ---------------------------------------------------------------------------
extern "C" void kernel_launch(void* const* d_in, const int* in_sizes, int n_in,
                              void* d_out, int out_size, void* d_ws, size_t ws_size,
                              hipStream_t stream) {
    const float* u   = (const float*)d_in[0]; // (8, 4096, 512)
    const float* lar = (const float*)d_in[1]; // (64,)
    const float* lai = (const float*)d_in[2]; // (64,)
    const float* Bw  = (const float*)d_in[3]; // (128, 512)
    const float* Cw  = (const float*)d_in[4]; // (512, 128)
    const float* Dw  = (const float*)d_in[5]; // (512, 512)
    float* y = (float*)d_out;                 // (8, 4096, 512)

    char* ws = (char*)d_ws;
    const size_t BU_BYTES  = (size_t)M_TOTAL * NPROJ * sizeof(float);            // 16 MB
    const size_t F_BYTES   = (size_t)BATCH * D_STATE * NCHUNK * sizeof(float2);  // 128 KB
    const size_t BW_ELEMS  = (size_t)NPROJ * D_MODEL;    // 65536
    const size_t CW_ELEMS  = (size_t)D_MODEL * NPROJ;    // 65536
    const size_t DW_ELEMS  = (size_t)D_MODEL * D_MODEL;  // 262144

    float*  Bu   = (float*)(ws);
    float*  H    = (float*)(ws + BU_BYTES);
    float2* F    = (float2*)(ws + 2 * BU_BYTES);
    float2* Hpre = (float2*)(ws + 2 * BU_BYTES + F_BYTES);
    __bf16* Bwb  = (__bf16*)(ws + 2 * BU_BYTES + 2 * F_BYTES);
    __bf16* Cwb  = Bwb + BW_ELEMS;
    __bf16* Dwb  = Cwb + CW_ELEMS;

    // One-time weight conversion (tiny; stays resident in L2).
    s4d_cvt_bf16<<<(int)(BW_ELEMS / 4 + 255) / 256, 256, 0, stream>>>(Bw, Bwb, (int)(BW_ELEMS / 4));
    s4d_cvt_bf16<<<(int)(CW_ELEMS / 4 + 255) / 256, 256, 0, stream>>>(Cw, Cwb, (int)(CW_ELEMS / 4));
    s4d_cvt_bf16<<<(int)(DW_ELEMS / 4 + 255) / 256, 256, 0, stream>>>(Dw, Dwb, (int)(DW_ELEMS / 4));

    s4d_proj_in<<<M_TOTAL / 128, 256, 0, stream>>>(u, Bwb, Bu);
    s4d_scan_local<<<BATCH * NCHUNK, D_STATE, 0, stream>>>(lar, lai, Bu, H, F);
    s4d_scan_combine<<<BATCH, D_STATE, 0, stream>>>(lar, lai, F, Hpre);
    s4d_scan_fix<<<BATCH * NCHUNK, D_STATE, 0, stream>>>(lar, lai, Hpre, H);
    s4d_proj_out<<<dim3(M_TOTAL / 128, D_MODEL / 128), 256, 0, stream>>>(H, u, Cwb, Dwb, y);
}
